// SMLP_eg_48404281426526
// MI455X (gfx1250) — matrix operand order repaired and checked by reading, not compiled, not run on hardware
//
#include <hip/hip_runtime.h>
#include <hip/hip_bf16.h>

// ---------------------------------------------------------------------------
// Types for CDNA5 WMMA (wave32, 16x16x32 f16 -> f32) and TDM descriptors
// ---------------------------------------------------------------------------
typedef __attribute__((ext_vector_type(16))) _Float16 v16h;
typedef __attribute__((ext_vector_type(8)))  _Float16 v8h;
typedef __attribute__((ext_vector_type(8)))  float    v8f;
typedef __attribute__((ext_vector_type(4)))  unsigned int tdm_v4u;
typedef __attribute__((ext_vector_type(8)))  int          tdm_v8i;
typedef __attribute__((ext_vector_type(4)))  int          tdm_v4i;

#define TT 100
#define BB 4096
#define MM (TT * BB)          // 409600 rows in the time-flattened GEMMs
#define LDSB_LD 264           // 256 halfs + 8-half pad per row (bank-conflict-free)

// ---------------------------------------------------------------------------
// Fragment loaders: contiguous k order (element s of lane half h holds k = 16h + s), same for A and B
// ---------------------------------------------------------------------------
static __device__ inline v16h load_frag_a(const _Float16* __restrict__ base,
                                          int lda, int m0, int k0, int lane) {
    const int row = m0 + (lane & 15);
    const int kb  = k0 + ((lane >> 4) << 4);
    const _Float16* p = base + (size_t)row * lda + kb;
    v8h lo = *(const v8h*)(p);        // K kb .. kb+7
    v8h hi = *(const v8h*)(p + 8);    // K kb+8 .. kb+15
    return __builtin_shufflevector(lo, hi, 0,1,2,3,4,5,6,7,8,9,10,11,12,13,14,15);
}

static __device__ inline v16h load_frag_b(const _Float16* __restrict__ w,
                                          int ldb, int n0, int k0, int lane) {
    const int nn = n0 + (lane & 15);
    const int kb = k0 + ((lane >> 4) << 4);
    const _Float16* p = w + (size_t)nn * ldb + kb;
    v8h lo = *(const v8h*)(p);
    v8h hi = *(const v8h*)(p + 8);
    return __builtin_shufflevector(lo, hi, 0,1,2,3,4,5,6,7,8,9,10,11,12,13,14,15);
}

static __device__ inline v8f wmma_f16(v16h a, v16h b, v8f c) {
    return __builtin_amdgcn_wmma_f32_16x16x32_f16(false, a, false, b,
                                                  (short)0, c, false, false);
}

// ---------------------------------------------------------------------------
// TDM: DMA a [64 x 256] f16 weight slab into LDS; TDM inserts a 16B pad every
// 512B row (pad_interval=6 -> 512B, pad_amount=3 -> 4 DWORDs) => LDS row
// stride 528B = 132 DWORDs (4-bank rotation/row, conflict-free B-frag reads).
// ---------------------------------------------------------------------------
static __device__ inline void tdm_load_slab(unsigned int lds_off,
                                            const _Float16* gsrc) {
    unsigned long long ga = (unsigned long long)(uintptr_t)gsrc;
    tdm_v4u g0 = { 1u,                                    // count=1, user mode
                   lds_off,                               // LDS byte address
                   (unsigned int)(ga & 0xFFFFFFFFull),    // global_addr[31:0]
                   (unsigned int)((ga >> 32) & 0x01FFFFFFull) | (2u << 30) }; // type=2
    tdm_v8i g1 = { 0x07910000,   // data_size=2B, pad_en, pad_interval=512B, pad=4DW
                   0x01000000,   // tensor_dim0 = 256
                   0x00400000,   // tensor_dim1 = 64
                   0x01000000,   // tile_dim0   = 256
                   64,           // tile_dim1   = 64
                   256,          // tensor_dim0_stride = 256 elements
                   0, 0 };
    tdm_v4i z4 = { 0, 0, 0, 0 };
#if __has_include(<hip/amd_detail/amd_gfx1250_TDM.h>)
    tdm_v8i z8 = { 0, 0, 0, 0, 0, 0, 0, 0 };
    __builtin_amdgcn_tensor_load_to_lds(g0, g1, z4, z4, z8, 0);
#else
    __builtin_amdgcn_tensor_load_to_lds(g0, g1, z4, z4, 0);
#endif
}

// ---------------------------------------------------------------------------
// Generic wave-tiled GEMM: C[MM x N] = A[MM x K](f16) * W[N x K]^T(f16) + bias
// One wave owns a 16 x (16*NT) tile; block = 4 waves stacked along M.
// OutT: float (final layer) or _Float16 (intermediates).
// SWZ:  store each 16x16 tile in fragment order [lane][r] (16B v8h per lane,
//       fully coalesced) for consumers that re-load C fragments (rec_step).
// ---------------------------------------------------------------------------
template <int NT, typename OutT, bool SWZ = false>
__global__ __launch_bounds__(128) void gemm_f16(const _Float16* __restrict__ A,
                                                const _Float16* __restrict__ Bw,
                                                OutT* __restrict__ C,
                                                const float* __restrict__ bias,
                                                int K, int lda, int ldb, int ldc,
                                                int Nvalid) {
    const int lane = threadIdx.x & 31;
    const int wave = threadIdx.x >> 5;
    const int m0   = (blockIdx.x * 4 + wave) * 16;
    const int n0   = blockIdx.y * (16 * NT);
    const int col  = lane & 15;
    const int hf   = lane >> 4;

    v8f acc[NT];
#pragma unroll
    for (int j = 0; j < NT; ++j) {
        float b = bias ? bias[n0 + j * 16 + col] : 0.0f;
#pragma unroll
        for (int r = 0; r < 8; ++r) acc[j][r] = b;
    }

    for (int k0 = 0; k0 < K; k0 += 32) {
        v16h af = load_frag_a(A, lda, m0, k0, lane);
#pragma unroll
        for (int j = 0; j < NT; ++j) {
            v16h bf = load_frag_b(Bw, ldb, n0 + j * 16, k0, lane);
            acc[j] = wmma_f16(af, bf, acc[j]);
        }
    }

#pragma unroll
    for (int j = 0; j < NT; ++j) {
        if constexpr (SWZ) {
            // tile index = (m0/16)*(N/16) + n0/16 + j ; 256 elems per tile
            size_t tb = ((size_t)(m0 >> 4) * (ldc >> 4) + (n0 >> 4) + j) * 256
                        + (size_t)lane * 8;
            v8h hv;
#pragma unroll
            for (int r = 0; r < 8; ++r) hv[r] = (_Float16)acc[j][r];
            *(v8h*)((_Float16*)C + tb) = hv;
        } else {
            int n = n0 + j * 16 + col;
            if (n < Nvalid) {
#pragma unroll
                for (int r = 0; r < 8; ++r) {
                    int m = m0 + r + hf * 8;
                    C[(size_t)m * ldc + n] = (OutT)acc[j][r];
                }
            }
        }
    }
}

// ---------------------------------------------------------------------------
// One recurrent timestep, fused: i = a_t + y_{t-1} @ Whh^T ; LIF update.
// M=4096 (B), N=256, K=256. Wave tile 16x64 -> 32 WMMAs per wave.
// Whh slab staged to LDS via TDM; a_t and vstate use the tile-swizzled
// fragment layout, so prologue/epilogue are coalesced b128 accesses.
// ---------------------------------------------------------------------------
__global__ __launch_bounds__(128) void rec_step(const _Float16* __restrict__ yprev,
                                                const _Float16* __restrict__ a_t,
                                                const _Float16* __restrict__ Whh,
                                                float* __restrict__ vstate,
                                                _Float16* __restrict__ yout) {
    __shared__ _Float16 sB[64 * LDSB_LD];

    const int lane = threadIdx.x & 31;
    const int wave = threadIdx.x >> 5;
    const int m0   = (blockIdx.x * 4 + wave) * 16;
    const int n0   = blockIdx.y * 64;
    const int col  = lane & 15;
    const int hf   = lane >> 4;
    const int mt   = m0 >> 4;          // row-tile index within the 4096-batch

    if (wave == 0) {
        tdm_load_slab((unsigned int)(uintptr_t)&sB[0], Whh + (size_t)n0 * 256);
        __builtin_amdgcn_s_wait_tensorcnt(0);
    }
    __syncthreads();

    // C-init from tile-swizzled f16 drive: one coalesced 16B load per tile.
    v8f acc[4];
#pragma unroll
    for (int j = 0; j < 4; ++j) {
        size_t tb = ((size_t)mt * 16 + (n0 >> 4) + j) * 256 + (size_t)lane * 8;
        v8h av = *(const v8h*)(a_t + tb);
#pragma unroll
        for (int r = 0; r < 8; ++r) acc[j][r] = (float)av[r];
    }

#pragma unroll
    for (int k0 = 0; k0 < 256; k0 += 32) {
        v16h af = load_frag_a(yprev, 256, m0, k0, lane);
#pragma unroll
        for (int j = 0; j < 4; ++j) {
            const int nn = j * 16 + col;
            const int kb = k0 + ((lane >> 4) << 4);
            const _Float16* p = &sB[nn * LDSB_LD + kb];
            v8h lo = *(const v8h*)(p);
            v8h hi = *(const v8h*)(p + 8);
            v16h bf = __builtin_shufflevector(lo, hi,
                        0,1,2,3,4,5,6,7,8,9,10,11,12,13,14,15);
            acc[j] = wmma_f16(af, bf, acc[j]);
        }
    }

    // LIF epilogue: vstate in tile-swizzled layout (coalesced 32B v8f r/w);
    // spikes stored row-major (feeds next step's A fragments and GEMM3).
#pragma unroll
    for (int j = 0; j < 4; ++j) {
        size_t tb = ((size_t)mt * 16 + (n0 >> 4) + j) * 256 + (size_t)lane * 8;
        v8f vv = *(v8f*)(vstate + tb);
#pragma unroll
        for (int r = 0; r < 8; ++r) {
            float v = vv[r];
            v += (acc[j][r] - v) * 0.5f;           // TAU = 2
            float s = (v >= 1.0f) ? 1.0f : 0.0f;   // VTH = 1
            vv[r] = v * (1.0f - s);                // hard reset
            int m = m0 + r + hf * 8;
            int n = n0 + j * 16 + col;
            yout[(size_t)m * 256 + n] = (_Float16)s;
        }
        *(v8f*)(vstate + tb) = vv;
    }
}

// ---------------------------------------------------------------------------
// Elementwise LIF scan over T for non-recurrent layers (parallel over B*D).
// ---------------------------------------------------------------------------
__global__ void lif_scan(const _Float16* __restrict__ h, _Float16* __restrict__ s,
                         int TD, int T) {
    int i = blockIdx.x * blockDim.x + threadIdx.x;
    if (i >= TD) return;
    float v = 0.0f;
    for (int t = 0; t < T; ++t) {
        float ht = (float)h[(size_t)t * TD + i];
        v += (ht - v) * 0.5f;
        float sp = (v >= 1.0f) ? 1.0f : 0.0f;
        v *= (1.0f - sp);
        s[(size_t)t * TD + i] = (_Float16)sp;
    }
}

// f32 [srcRows x kin] -> f16 [dstRows x kout], zero padded
__global__ void cvt_pad(const float* __restrict__ src, _Float16* __restrict__ dst,
                        int srcRows, int dstRows, int kin, int kout) {
    long long idx = (long long)blockIdx.x * blockDim.x + threadIdx.x;
    long long total = (long long)dstRows * kout;
    if (idx >= total) return;
    int r = (int)(idx / kout);
    int k = (int)(idx % kout);
    float v = (r < srcRows && k < kin) ? src[(size_t)r * kin + k] : 0.0f;
    dst[idx] = (_Float16)v;
}

// f32 slice [rows x cols] at column offset (Whh = W_rc[:, 256:]) -> f16
__global__ void cvt_slice(const float* __restrict__ src, _Float16* __restrict__ dst,
                          int rows, int srcLd, int colOff, int cols) {
    int idx = blockIdx.x * blockDim.x + threadIdx.x;
    if (idx >= rows * cols) return;
    int r = idx / cols, c = idx % cols;
    dst[idx] = (_Float16)src[(size_t)r * srcLd + colOff + c];
}

// Wc[p,i] = sum_o W_rc[p,o] * W2[o,i]  (fold layer-2 into the input drive)
__global__ void wc_kernel(const float* __restrict__ Wrc, const float* __restrict__ W2,
                          _Float16* __restrict__ Wc) {
    int idx = blockIdx.x * blockDim.x + threadIdx.x;
    if (idx >= 256 * 64) return;
    int p = idx >> 6, i = idx & 63;
    float acc = 0.0f;
    for (int o = 0; o < 256; ++o)
        acc += Wrc[(size_t)p * 512 + o] * W2[(size_t)o * 64 + i];
    Wc[idx] = (_Float16)acc;
}

// ---------------------------------------------------------------------------
// Host-side orchestration
// ---------------------------------------------------------------------------
extern "C" void kernel_launch(void* const* d_in, const int* in_sizes, int n_in,
                              void* d_out, int out_size, void* d_ws, size_t ws_size,
                              hipStream_t stream) {
    (void)in_sizes; (void)n_in; (void)out_size; (void)ws_size;

    const float* x   = (const float*)d_in[0];  // [100,4096,40]
    const float* W1  = (const float*)d_in[1];  // [64,40]
    const float* W2  = (const float*)d_in[2];  // [256,64]
    const float* Wrc = (const float*)d_in[3];  // [256,512]
    const float* brc = (const float*)d_in[4];  // [256]
    const float* W3  = (const float*)d_in[5];  // [128,256]
    const float* W4  = (const float*)d_in[6];  // [35,128]
    float* out = (float*)d_out;                // [100,4096,35]

    char* ws = (char*)d_ws;
    size_t off = 0;
    auto alloc = [&](size_t bytes) -> void* {
        void* p = ws + off;
        off = (off + bytes + 255) & ~(size_t)255;
        return p;
    };

    _Float16* xpad = (_Float16*)alloc((size_t)MM * 64 * 2);   // x padded K40->64
    _Float16* w1p  = (_Float16*)alloc(64 * 64 * 2);
    _Float16* wcm  = (_Float16*)alloc(256 * 64 * 2);          // Wx @ W2
    _Float16* whh  = (_Float16*)alloc(256 * 256 * 2);         // W_rc[:,256:]
    _Float16* w3p  = (_Float16*)alloc(128 * 256 * 2);
    _Float16* w4p  = (_Float16*)alloc(48 * 128 * 2);          // rows padded 35->48
    _Float16* h1   = (_Float16*)alloc((size_t)MM * 64 * 2);   // f16 intermediates
    _Float16* s1   = (_Float16*)alloc((size_t)MM * 64 * 2);
    _Float16* abuf = (_Float16*)alloc((size_t)MM * 256 * 2);  // drive a (swizzled f16)
    _Float16* s2   = (_Float16*)alloc((size_t)(TT + 1) * BB * 256 * 2); // slot0=y_{-1}
    float*    v2   = (float*)alloc((size_t)BB * 256 * 4);     // v state (swizzled)
    _Float16* h3   = (_Float16*)alloc((size_t)MM * 128 * 2);
    _Float16* s3   = (_Float16*)alloc((size_t)MM * 128 * 2);

    const int thr = 256;

    // ---- weight / input preparation ----
    {
        long long total = (long long)MM * 64;
        cvt_pad<<<(unsigned)((total + thr - 1) / thr), thr, 0, stream>>>(x, xpad, MM, MM, 40, 64);
    }
    cvt_pad<<<(64 * 64 + thr - 1) / thr, thr, 0, stream>>>(W1, w1p, 64, 64, 40, 64);
    cvt_slice<<<(256 * 256 + thr - 1) / thr, thr, 0, stream>>>(Wrc, whh, 256, 512, 256, 256);
    cvt_pad<<<(128 * 256 + thr - 1) / thr, thr, 0, stream>>>(W3, w3p, 128, 128, 256, 256);
    cvt_pad<<<(48 * 128 + thr - 1) / thr, thr, 0, stream>>>(W4, w4p, 35, 48, 128, 128);
    wc_kernel<<<(256 * 64 + thr - 1) / thr, thr, 0, stream>>>(Wrc, W2, wcm);

    // ---- layer 1: h1 = x @ W1^T (all T in parallel), then LIF scan ----
    gemm_f16<4, _Float16><<<dim3(MM / 64, 1), 128, 0, stream>>>(xpad, w1p, h1, nullptr,
                                                                64, 64, 64, 64, 64);
    lif_scan<<<(BB * 64 + thr - 1) / thr, thr, 0, stream>>>(h1, s1, BB * 64, TT);

    // ---- input drive: a = s1 @ (Wx*W2)^T + b, stored tile-swizzled ----
    gemm_f16<4, _Float16, true><<<dim3(MM / 64, 4), 128, 0, stream>>>(s1, wcm, abuf, brc,
                                                                      64, 64, 64, 256, 256);

    // ---- recurrent LIF: 100 sequential fused GEMM+LIF steps ----
    hipMemsetAsync(s2, 0, (size_t)BB * 256 * 2, stream);   // y_{-1} = 0
    hipMemsetAsync(v2, 0, (size_t)BB * 256 * 4, stream);   // v = 0
    for (int t = 0; t < TT; ++t) {
        rec_step<<<dim3(BB / 64, 4), 128, 0, stream>>>(
            s2 + (size_t)t * BB * 256,
            abuf + (size_t)t * BB * 256,
            whh, v2,
            s2 + (size_t)(t + 1) * BB * 256);
    }

    // ---- layer 3: h3 = s2 @ W3^T (all T), LIF scan ----
    gemm_f16<4, _Float16><<<dim3(MM / 64, 2), 128, 0, stream>>>(s2 + (size_t)BB * 256,
                                                                w3p, h3, nullptr,
                                                                256, 256, 256, 128, 128);
    lif_scan<<<(BB * 128 + thr - 1) / thr, thr, 0, stream>>>(h3, s3, BB * 128, TT);

    // ---- layer 4: out = s3 @ W4^T, N padded to 48, stores predicated to 35 ----
    gemm_f16<3, float><<<dim3(MM / 64, 1), 128, 0, stream>>>(s3, w4p, out, nullptr,
                                                             128, 128, 128, 35, 35);
}
